// selfattention_4810363372891
// MI455X (gfx1250) — matrix operand order repaired and checked by reading, not compiled, not run on hardware
//
#include <hip/hip_runtime.h>
#include <hip/hip_bf16.h>

// ---------------------------------------------------------------------------
// Self-attention, N=8192 tokens, C=256 channels, fp32 in/out.
// Stage 1: QKV projection GEMM with v_wmma_f32_16x16x32_f16 (V stored
//          transposed so the attention loop never transposes).
// Stage 2: flash attention, double-buffered LDS tiles fed by
//          global_load_async_to_lds_b128 (ASYNCcnt) when available.
// ---------------------------------------------------------------------------

#define N_TOK 8192
#define CDIM  256

typedef _Float16 v16h __attribute__((ext_vector_type(16)));
typedef _Float16 v8h  __attribute__((ext_vector_type(8)));
typedef float    v8f  __attribute__((ext_vector_type(8)));
typedef float    v4f  __attribute__((ext_vector_type(4)));

#if defined(__HIP_DEVICE_COMPILE__) && \
    __has_builtin(__builtin_amdgcn_global_load_async_to_lds_b128)
#define HAVE_ASYNC_LDS 1
typedef int v4i __attribute__((vector_size(16)));
typedef __attribute__((address_space(1))) v4i as1_v4i;  // global int4
typedef __attribute__((address_space(3))) v4i as3_v4i;  // LDS int4
#else
#define HAVE_ASYNC_LDS 0
#endif

static __device__ __forceinline__ v16h cat8(v8h lo, v8h hi) {
  v16h r;
#pragma unroll
  for (int i = 0; i < 8; ++i) { r[i] = lo[i]; r[i + 8] = hi[i]; }
  return r;
}

// load 8 consecutive fp32, convert to f16
static __device__ __forceinline__ v8h cvt8(const float* __restrict__ p) {
  v4f a = *(const v4f*)p;
  v4f b = *(const v4f*)(p + 4);
  v8h r;
#pragma unroll
  for (int i = 0; i < 4; ++i) { r[i] = (_Float16)a[i]; r[i + 4] = (_Float16)b[i]; }
  return r;
}

static __device__ __forceinline__ v8f wmma16(v16h a, v16h b, v8f c) {
  return __builtin_amdgcn_wmma_f32_16x16x32_f16(
      /*neg_a=*/false, a, /*neg_b=*/false, b,
      /*c_mod=*/(short)0, c, /*reuse_a=*/false, /*reuse_b=*/false);
}

// 16-byte global -> LDS copy; async (no VGPR staging) when the toolchain has it
static __device__ __forceinline__ void copy16(const _Float16* __restrict__ g,
                                              _Float16* l) {
#if HAVE_ASYNC_LDS
  __builtin_amdgcn_global_load_async_to_lds_b128((as1_v4i*)g, (as3_v4i*)l,
                                                 0, 0);
#else
  *(uint4*)l = *(const uint4*)g;
#endif
}

static __device__ __forceinline__ void wait_async0() {
#if HAVE_ASYNC_LDS
#if __has_builtin(__builtin_amdgcn_s_wait_asynccnt)
  __builtin_amdgcn_s_wait_asynccnt(0);
#else
  asm volatile("s_wait_asynccnt 0" ::: "memory");
#endif
#endif
}

// ---------------------------------------------------------------------------
// Kernel 1: q/k/v = x @ W^T   (x fp32 [N,C], W fp32 [C,C] row-major)
// One wave -> 16(M) x 64(N) tile, 8 K-steps of 32. V goes out transposed.
// ---------------------------------------------------------------------------
__global__ __launch_bounds__(128) void qkv_gemm_kernel(
    const float* __restrict__ x,
    const float* __restrict__ Wq, const float* __restrict__ Wk,
    const float* __restrict__ Wv,
    _Float16* __restrict__ qh, _Float16* __restrict__ kh,
    _Float16* __restrict__ vht) {
  const int lane = threadIdx.x & 31;
  const int wave = threadIdx.x >> 5;
  const int half = lane >> 4;
  const int l16  = lane & 15;

  const int mbase = blockIdx.x * 16;
  const int which = blockIdx.y;
  const float* __restrict__ W = (which == 0) ? Wq : (which == 1) ? Wk : Wv;

  const int cb = wave * 64;
  const float* __restrict__ xrow = x + (size_t)(mbase + l16) * CDIM;

  v8f acc[4];
#pragma unroll
  for (int j = 0; j < 4; ++j)
#pragma unroll
    for (int g = 0; g < 8; ++g) acc[j][g] = 0.0f;

#pragma unroll
  for (int kc = 0; kc < 8; ++kc) {
    const int k0 = kc * 32;
    v16h a = cat8(cvt8(xrow + k0 + half * 16),
                  cvt8(xrow + k0 + half * 16 + 8));
#pragma unroll
    for (int j = 0; j < 4; ++j) {
      const int col = cb + j * 16 + l16;
      const float* __restrict__ wp = W + (size_t)col * CDIM + k0 + half * 16;
      v16h b = cat8(cvt8(wp), cvt8(wp + 8));
      acc[j] = wmma16(a, b, acc[j]);
    }
  }

  // C/D layout: element (M = g + 8*half, N = l16) lives in VGPR g.
  if (which == 2) {
    // store V transposed: vht[dim][token] -> attention B operand is contiguous
#pragma unroll
    for (int j = 0; j < 4; ++j)
#pragma unroll
      for (int g = 0; g < 8; ++g) {
        const int M = g + 8 * half;
        vht[(size_t)(cb + j * 16 + l16) * N_TOK + mbase + M] =
            (_Float16)acc[j][g];
      }
  } else {
    _Float16* __restrict__ Y = (which == 0) ? qh : kh;
#pragma unroll
    for (int j = 0; j < 4; ++j)
#pragma unroll
      for (int g = 0; g < 8; ++g) {
        const int M = g + 8 * half;
        Y[(size_t)(mbase + M) * CDIM + cb + j * 16 + l16] =
            (_Float16)acc[j][g];
      }
  }
}

// issue one tile's worth of LDS copies (K row-major 32x256, Vt 256x32)
static __device__ __forceinline__ void load_tiles(
    _Float16* ksDst, _Float16* vtDst, const _Float16* __restrict__ kh,
    const _Float16* __restrict__ vht, int key0, int tid) {
#pragma unroll
  for (int i = 0; i < 4; ++i) {
    const int chunk = tid + i * 256;  // 1024 chunks of 8 halves
    copy16(kh + (size_t)key0 * CDIM + chunk * 8, ksDst + chunk * 8);
  }
#pragma unroll
  for (int i = 0; i < 4; ++i) {
    const int chunk = tid + i * 256;  // dim = chunk/4, key-offset = (chunk%4)*8
    const int dim = chunk >> 2;
    const int ko  = (chunk & 3) * 8;
    copy16(vht + (size_t)dim * N_TOK + key0 + ko, vtDst + chunk * 8);
  }
}

// ---------------------------------------------------------------------------
// Kernel 2: flash attention. Block = 256 threads = 8 waves; each wave owns
// 16 query rows. Key tiles of 32, double-buffered in LDS.
// ---------------------------------------------------------------------------
__global__ __launch_bounds__(256) void flash_attn_kernel(
    const _Float16* __restrict__ qh, const _Float16* __restrict__ kh,
    const _Float16* __restrict__ vht, float* __restrict__ out) {
  __shared__ __align__(16) _Float16 KsBuf[2][32 * CDIM];   // 2 x 16KB
  __shared__ __align__(16) _Float16 VtBuf[2][CDIM * 32];   // 2 x 16KB
  __shared__ __align__(16) _Float16 Pw[8][16 * 32];        // 8KB

  const int tid  = threadIdx.x;
  const int lane = tid & 31;
  const int wave = tid >> 5;
  const int half = lane >> 4;
  const int l16  = lane & 15;

  const int qrow0 = blockIdx.x * 128 + wave * 16;
  const int NT    = N_TOK / 32;

  // prologue: start tile 0 transfer, overlap with Q preload
  load_tiles(KsBuf[0], VtBuf[0], kh, vht, 0, tid);

  v16h Aq[8];
#pragma unroll
  for (int kc = 0; kc < 8; ++kc) {
    const _Float16* __restrict__ qp =
        qh + (size_t)(qrow0 + l16) * CDIM + kc * 32 + half * 16;
    Aq[kc] = cat8(*(const v8h*)qp, *(const v8h*)(qp + 8));
  }

  float m[8], lsum[8];
  v8f O[16];
#pragma unroll
  for (int g = 0; g < 8; ++g) { m[g] = -1e30f; lsum[g] = 0.0f; }
#pragma unroll
  for (int c = 0; c < 16; ++c)
#pragma unroll
    for (int g = 0; g < 8; ++g) O[c][g] = 0.0f;

  const float scale = 0.0625f;  // 1/sqrt(256)

  wait_async0();
  __syncthreads();

  for (int kt = 0; kt < NT; ++kt) {
    const int cur = kt & 1;
    // prefetch next tile into the other buffer (overlaps this tile's math;
    // buf[cur^1] was last read before the previous barrier)
    if (kt + 1 < NT)
      load_tiles(KsBuf[cur ^ 1], VtBuf[cur ^ 1], kh, vht, (kt + 1) * 32, tid);

    const _Float16* Ks = KsBuf[cur];
    const _Float16* Vt = VtBuf[cur];

    // ---- S = Q @ K^T : keys 0-15 in S0, keys 16-31 in S1 ----
    v8f S0, S1;
#pragma unroll
    for (int g = 0; g < 8; ++g) { S0[g] = 0.0f; S1[g] = 0.0f; }
#pragma unroll
    for (int kc = 0; kc < 8; ++kc) {
      v16h B0 = *(const v16h*)&Ks[l16 * CDIM + kc * 32 + half * 16];
      v16h B1 = *(const v16h*)&Ks[(l16 + 16) * CDIM + kc * 32 + half * 16];
      S0 = wmma16(Aq[kc], B0, S0);
      S1 = wmma16(Aq[kc], B1, S1);
    }

    // ---- online softmax; row = (g, half); cols across each 16-lane half ----
    float al[8], P0[8], P1[8];
#pragma unroll
    for (int g = 0; g < 8; ++g) {
      S0[g] *= scale; S1[g] *= scale;
      float v = fmaxf(S0[g], S1[g]);
      v = fmaxf(v, __shfl_xor(v, 1, 32));
      v = fmaxf(v, __shfl_xor(v, 2, 32));
      v = fmaxf(v, __shfl_xor(v, 4, 32));
      v = fmaxf(v, __shfl_xor(v, 8, 32));
      const float nm = fmaxf(m[g], v);
      al[g] = __expf(m[g] - nm);
      m[g] = nm;
    }
#pragma unroll
    for (int g = 0; g < 8; ++g) {
      P0[g] = __expf(S0[g] - m[g]);
      P1[g] = __expf(S1[g] - m[g]);
      float s = P0[g] + P1[g];
      s += __shfl_xor(s, 1, 32);
      s += __shfl_xor(s, 2, 32);
      s += __shfl_xor(s, 4, 32);
      s += __shfl_xor(s, 8, 32);
      lsum[g] = lsum[g] * al[g] + s;
    }
#pragma unroll
    for (int c = 0; c < 16; ++c)
#pragma unroll
      for (int g = 0; g < 8; ++g) O[c][g] *= al[g];

    // ---- P: C-layout -> A-layout via per-wave LDS bounce ----
    _Float16* pw = &Pw[wave][0];
#pragma unroll
    for (int g = 0; g < 8; ++g) {
      const int M = g + 8 * half;
      pw[M * 32 + l16]      = (_Float16)P0[g];
      pw[M * 32 + l16 + 16] = (_Float16)P1[g];
    }
    asm volatile("s_wait_dscnt 0" ::: "memory");  // wave-internal cross-lane dep
    const _Float16* pr = pw + l16 * 32 + half * 16;
    v16h Ap = cat8(*(const v8h*)pr, *(const v8h*)(pr + 8));

    // ---- O += P @ V over 16 column chunks ----
#pragma unroll
    for (int c = 0; c < 16; ++c) {
      v16h Bv = *(const v16h*)&Vt[(c * 16 + l16) * 32 + half * 16];
      O[c] = wmma16(Ap, Bv, O[c]);
    }

    wait_async0();   // next tile's async transfers landed
    __syncthreads(); // everyone done reading buf[cur] before it's overwritten
  }

  // ---- epilogue: out = O / l ----
#pragma unroll
  for (int c = 0; c < 16; ++c)
#pragma unroll
    for (int g = 0; g < 8; ++g) {
      const int M = g + 8 * half;
      out[(size_t)(qrow0 + M) * CDIM + c * 16 + l16] = O[c][g] / lsum[g];
    }
}

// ---------------------------------------------------------------------------
extern "C" void kernel_launch(void* const* d_in, const int* in_sizes, int n_in,
                              void* d_out, int out_size, void* d_ws,
                              size_t ws_size, hipStream_t stream) {
  (void)in_sizes; (void)n_in; (void)out_size; (void)ws_size;
  const float* x  = (const float*)d_in[0];
  const float* Wq = (const float*)d_in[1];
  const float* Wk = (const float*)d_in[2];
  const float* Wv = (const float*)d_in[3];
  float* out = (float*)d_out;

  // workspace: q, k row-major + v transposed, all f16 (12 MB)
  _Float16* qh  = (_Float16*)d_ws;
  _Float16* kh  = qh + (size_t)N_TOK * CDIM;
  _Float16* vht = kh + (size_t)N_TOK * CDIM;

  qkv_gemm_kernel<<<dim3(N_TOK / 16, 3), 128, 0, stream>>>(
      x, Wq, Wk, Wv, qh, kh, vht);
  flash_attn_kernel<<<dim3(N_TOK / 128), 256, 0, stream>>>(qh, kh, vht, out);
}
